// LSTM_Decoder_28810640621653
// MI455X (gfx1250) — matrix operand order repaired and checked by reading, not compiled, not run on hardware
//
#include <hip/hip_runtime.h>
#include <hip/hip_bf16.h>

#define VOCAB  32000
#define EMBED  512
#define HIDDEN 1024
#define BATCH  32
#define SEQ    48

typedef __bf16 bf16_t;
typedef __attribute__((ext_vector_type(16))) __bf16 v16bf;
typedef __attribute__((ext_vector_type(8)))  float  v8f;

// ---------------------------------------------------------------------------
// WMMA fragment loaders (bf16, 16x16x32, per CDNA5 ISA VGPR layouts)
// A: 16(M) x 32(K), row-major source with leading dim lda (elements).
//   lane<16 : row=lane,    K chunks [k0..k0+7] and [k0+16..k0+23]
//   lane>=16: row=lane-16, K chunks [k0+8..k0+15] and [k0+24..k0+31]
// ---------------------------------------------------------------------------
__device__ __forceinline__ v16bf load_a_frag(const bf16_t* __restrict__ A,
                                             int lda, int mbase, int k0, int lane) {
    const bf16_t* p = A + (size_t)(mbase + (lane & 15)) * lda + k0 + ((lane & 16) ? 8 : 0);
    union { uint4 u[2]; v16bf v; } t;
    t.u[0] = *(const uint4*)(p);        // 8 bf16
    t.u[1] = *(const uint4*)(p + 16);   // 8 bf16, +16 elements
    return t.v;
}

// B: 32(K) x 16(N) where B = W^T, W row-major [N][K] with leading dim ldb.
//   column n = W row n (contiguous K).
//   lane<16 : n=lane,    K = [k0..k0+7],[k0+16..k0+23] ; lane>=16: n=lane-16, K = [k0+8..k0+15],[k0+24..k0+31]
__device__ __forceinline__ v16bf load_b_frag(const bf16_t* __restrict__ W,
                                             int ldb, int nbase, int k0, int lane) {
    const bf16_t* p = W + (size_t)(nbase + (lane & 15)) * ldb + k0 + ((lane & 16) ? 8 : 0);
    union { uint4 u[2]; v16bf v; } t;
    t.u[0] = *(const uint4*)(p);
    t.u[1] = *(const uint4*)(p + 16);
    return t.v;
}

__device__ __forceinline__ v8f wmma_bf16(v16bf a, v16bf b, v8f c) {
    return __builtin_amdgcn_wmma_f32_16x16x32_bf16(false, a, false, b, (short)0, c,
                                                   false, false);
}

__device__ __forceinline__ float sigmoidf_(float x) {
    return 1.0f / (1.0f + __expf(-x));
}

// ---------------------------------------------------------------------------
// f32 -> bf16 conversion (weights cached in ws as bf16)
// ---------------------------------------------------------------------------
__global__ void convert_f32_bf16_kernel(const float* __restrict__ src,
                                        bf16_t* __restrict__ dst, int n) {
    int i = blockIdx.x * blockDim.x + threadIdx.x;
    if (i < n) dst[i] = (bf16_t)src[i];
}

// Gather embedding rows for all (t,b): Xall[t*32+b][0..511] = bf16(emb[inputs[b][t]])
__global__ __launch_bounds__(256) void gather_embed_kernel(const int* __restrict__ inputs,
                                                           const float* __restrict__ emb,
                                                           bf16_t* __restrict__ Xall) {
    int row = blockIdx.x;                 // t*BATCH + b
    int t = row >> 5, b = row & 31;
    int idx = inputs[b * SEQ + t];
    const float* src = emb + (size_t)idx * EMBED;
    bf16_t* dst = Xall + (size_t)row * EMBED;
    for (int e = threadIdx.x; e < EMBED; e += blockDim.x) dst[e] = (bf16_t)src[e];
}

__global__ void init_state_kernel(const float* __restrict__ hidden,
                                  const float* __restrict__ cell,
                                  bf16_t* __restrict__ H0, float* __restrict__ Cst) {
    int i = blockIdx.x * blockDim.x + threadIdx.x;
    if (i < BATCH * HIDDEN) { H0[i] = (bf16_t)hidden[i]; Cst[i] = cell[i]; }
}

// ---------------------------------------------------------------------------
// One LSTM timestep. Grid (64 jtiles, 2 mtiles), block = 128 threads (4 waves).
// Wave g computes the 16x16 tile of gate g for (mtile, jtile), K = 1536
// (512 from x_t @ W_ih^T + 1024 from h @ W_hh^T). Tiles meet in LDS for the
// fused elementwise cell update.
// ---------------------------------------------------------------------------
__global__ __launch_bounds__(128) void lstm_step_kernel(
    const bf16_t* __restrict__ Xall,   // [S*B, E]
    const bf16_t* __restrict__ Wih,    // [4H, E]
    const bf16_t* __restrict__ Whh,    // [4H, H]
    const bf16_t* __restrict__ Hread,  // [B, H] bf16
    bf16_t* __restrict__ Hwrite,       // [B, H] bf16
    float* __restrict__ Cst,           // [B, H] f32
    bf16_t* __restrict__ Hall,         // [B*S, H] row b*SEQ+t
    const float* __restrict__ b_ih, const float* __restrict__ b_hh, int t) {
    __shared__ float gtile[4][16][16];
    const int lane = threadIdx.x & 31;
    const int gate = threadIdx.x >> 5;           // 0..3 = i,f,g,o
    const int jt = blockIdx.x;                   // 0..63
    const int mbase = blockIdx.y * 16;           // batch tile
    const int nrow = gate * HIDDEN + jt * 16;    // gate-row base in [0,4H)

    v8f acc = {};
    const bf16_t* Xrow = Xall + (size_t)t * BATCH * EMBED;
#pragma unroll 4
    for (int k = 0; k < EMBED; k += 32) {
        v16bf a = load_a_frag(Xrow, EMBED, mbase, k, lane);
        v16bf b = load_b_frag(Wih, EMBED, nrow, k, lane);
        acc = wmma_bf16(a, b, acc);
    }
#pragma unroll 4
    for (int k = 0; k < HIDDEN; k += 32) {
        v16bf a = load_a_frag(Hread, HIDDEN, mbase, k, lane);
        v16bf b = load_b_frag(Whh, HIDDEN, nrow, k, lane);
        acc = wmma_bf16(a, b, acc);
    }
    const int n = lane & 15;
    const int mofs = (lane >> 4) * 8;
#pragma unroll
    for (int r = 0; r < 8; ++r) gtile[gate][mofs + r][n] = acc[r];
    __syncthreads();

    // 256 cell updates, 128 threads -> 2 each
    for (int e = threadIdx.x; e < 256; e += 128) {
        int mm = e >> 4, jj = e & 15;
        int b = mbase + mm;
        int j = jt * 16 + jj;
        float gi = gtile[0][mm][jj] + b_ih[0 * HIDDEN + j] + b_hh[0 * HIDDEN + j];
        float gf = gtile[1][mm][jj] + b_ih[1 * HIDDEN + j] + b_hh[1 * HIDDEN + j];
        float gg = gtile[2][mm][jj] + b_ih[2 * HIDDEN + j] + b_hh[2 * HIDDEN + j];
        float go = gtile[3][mm][jj] + b_ih[3 * HIDDEN + j] + b_hh[3 * HIDDEN + j];
        float c_old = Cst[(size_t)b * HIDDEN + j];
        float c_new = sigmoidf_(gf) * c_old + sigmoidf_(gi) * tanhf(gg);
        float h_new = sigmoidf_(go) * tanhf(c_new);
        Cst[(size_t)b * HIDDEN + j] = c_new;
        bf16_t hb = (bf16_t)h_new;
        Hwrite[(size_t)b * HIDDEN + j] = hb;
        Hall[((size_t)b * SEQ + t) * HIDDEN + j] = hb;
    }
}

// ---------------------------------------------------------------------------
// Batched projection: out[r][v] = Hall[r] . fc_W[v] + fc_b[v]
// M=1536, N=32000, K=1024.
// Register-blocked: each wave computes a 32(M) x 64(N) tile = 8 WMMA tiles,
// so per k-step it loads 2 A frags + 4 B frags for 8 WMMAs (~22 FLOP/byte of
// cache traffic, 8 independent accumulator chains). Block = 4 waves (2M x 2N)
// -> 64 x 128 block tile. Grid (250, 24). fc_W (bf16, 64 MB) lives in L2.
// ---------------------------------------------------------------------------
__global__ __launch_bounds__(128) void logits_gemm_kernel(
    const bf16_t* __restrict__ Hall,   // [1536, 1024]
    const bf16_t* __restrict__ fcWb,   // [32000, 1024]
    const float* __restrict__ fc_b,
    float* __restrict__ out) {         // [1536, 32000]
    const int lane = threadIdx.x & 31;
    const int wave = threadIdx.x >> 5;                 // 0..3
    const int mbase = blockIdx.y * 64 + (wave & 1) * 32;
    const int nbase = blockIdx.x * 128 + (wave >> 1) * 64;

    v8f acc[2][4] = {};
#pragma unroll 2
    for (int k = 0; k < HIDDEN; k += 32) {
        v16bf a0 = load_a_frag(Hall, HIDDEN, mbase, k, lane);
        v16bf a1 = load_a_frag(Hall, HIDDEN, mbase + 16, k, lane);
        v16bf b0 = load_b_frag(fcWb, HIDDEN, nbase, k, lane);
        v16bf b1 = load_b_frag(fcWb, HIDDEN, nbase + 16, k, lane);
        v16bf b2 = load_b_frag(fcWb, HIDDEN, nbase + 32, k, lane);
        v16bf b3 = load_b_frag(fcWb, HIDDEN, nbase + 48, k, lane);
        acc[0][0] = wmma_bf16(a0, b0, acc[0][0]);
        acc[0][1] = wmma_bf16(a0, b1, acc[0][1]);
        acc[0][2] = wmma_bf16(a0, b2, acc[0][2]);
        acc[0][3] = wmma_bf16(a0, b3, acc[0][3]);
        acc[1][0] = wmma_bf16(a1, b0, acc[1][0]);
        acc[1][1] = wmma_bf16(a1, b1, acc[1][1]);
        acc[1][2] = wmma_bf16(a1, b2, acc[1][2]);
        acc[1][3] = wmma_bf16(a1, b3, acc[1][3]);
    }
    const int n = lane & 15;
    const int mofs = (lane >> 4) * 8;
#pragma unroll
    for (int j = 0; j < 4; ++j) {
        const int col = nbase + j * 16 + n;
        const float bias = fc_b[col];
#pragma unroll
        for (int i = 0; i < 2; ++i) {
            const int row0 = mbase + i * 16 + mofs;
#pragma unroll
            for (int r = 0; r < 8; ++r)
                out[(size_t)(row0 + r) * VOCAB + col] = acc[i][j][r] + bias;
        }
    }
}

// ---------------------------------------------------------------------------
extern "C" void kernel_launch(void* const* d_in, const int* in_sizes, int n_in,
                              void* d_out, int out_size, void* d_ws, size_t ws_size,
                              hipStream_t stream) {
    const int*   inputs = (const int*)d_in[0];     // [B, S]
    const float* hidden = (const float*)d_in[1];   // [B, H]
    const float* cell   = (const float*)d_in[2];   // [B, H]
    const float* emb    = (const float*)d_in[3];   // [V, E]
    const float* W_ih   = (const float*)d_in[4];   // [4H, E]
    const float* W_hh   = (const float*)d_in[5];   // [4H, H]
    const float* b_ih   = (const float*)d_in[6];   // [4H]
    const float* b_hh   = (const float*)d_in[7];   // [4H]
    const float* fc_W   = (const float*)d_in[8];   // [V, H]
    const float* fc_b   = (const float*)d_in[9];   // [V]
    float* out = (float*)d_out;

    char* ws = (char*)d_ws;
    const size_t N_WIH = (size_t)4 * HIDDEN * EMBED;    //  2,097,152
    const size_t N_WHH = (size_t)4 * HIDDEN * HIDDEN;   //  4,194,304
    const size_t N_FCW = (size_t)VOCAB * HIDDEN;        // 32,768,000
    size_t off = 0;
    bf16_t* Wih_b = (bf16_t*)(ws + off); off += N_WIH * 2;                 // 4 MB
    bf16_t* Whh_b = (bf16_t*)(ws + off); off += N_WHH * 2;                 // 8 MB
    bf16_t* fcW_b = (bf16_t*)(ws + off); off += N_FCW * 2;                 // 64 MB
    bf16_t* Xall  = (bf16_t*)(ws + off); off += (size_t)SEQ * BATCH * EMBED * 2;
    bf16_t* Hbuf0 = (bf16_t*)(ws + off); off += (size_t)BATCH * HIDDEN * 2;
    bf16_t* Hbuf1 = (bf16_t*)(ws + off); off += (size_t)BATCH * HIDDEN * 2;
    float*  Cst   = (float*)(ws + off);  off += (size_t)BATCH * HIDDEN * 4;
    bf16_t* Hall  = (bf16_t*)(ws + off); off += (size_t)BATCH * SEQ * HIDDEN * 2;
    if (off > ws_size) return;  // workspace too small; nothing safe to do

    // Stage 0: precision conversion + embedding gather + state init
    convert_f32_bf16_kernel<<<(int)((N_WIH + 255) / 256), 256, 0, stream>>>(W_ih, Wih_b, (int)N_WIH);
    convert_f32_bf16_kernel<<<(int)((N_WHH + 255) / 256), 256, 0, stream>>>(W_hh, Whh_b, (int)N_WHH);
    convert_f32_bf16_kernel<<<(int)((N_FCW + 255) / 256), 256, 0, stream>>>(fc_W, fcW_b, (int)N_FCW);
    gather_embed_kernel<<<SEQ * BATCH, 256, 0, stream>>>(inputs, emb, Xall);
    init_state_kernel<<<(BATCH * HIDDEN + 255) / 256, 256, 0, stream>>>(hidden, cell, Hbuf0, Cst);

    // Stage 1: sequential recurrence (stream order provides the t->t+1 dependency)
    bf16_t* Hr = Hbuf0;
    bf16_t* Hw = Hbuf1;
    for (int t = 0; t < SEQ; ++t) {
        lstm_step_kernel<<<dim3(HIDDEN / 16, BATCH / 16), 128, 0, stream>>>(
            Xall, Wih_b, Whh_b, Hr, Hw, Cst, Hall, b_ih, b_hh, t);
        bf16_t* tmp = Hr; Hr = Hw; Hw = tmp;
    }

    // Stage 2: one big projection GEMM over all (b,t) rows
    logits_gemm_kernel<<<dim3(VOCAB / 128, (BATCH * SEQ) / 64), 128, 0, stream>>>(
        Hall, fcW_b, fc_b, out);
}
